// StandardAttention_32152125178319
// MI455X (gfx1250) — matrix operand order repaired and checked by reading, not compiled, not run on hardware
//
#include <hip/hip_runtime.h>
#include <hip/hip_bf16.h>

// ---------------------------------------------------------------------------
// MI455X (gfx1250) attention layer: QKV proj + RoPE + GQA flash attn + out proj
// Matmuls on v_wmma_f32_16x16x32_f16 (wave32); A/V tiles staged into LDS with
// the Tensor Data Mover (tensor_load_to_lds + s_wait_tensorcnt).
// ---------------------------------------------------------------------------

#define B_    2
#define S_    2048
#define DIM_  4096
#define NH_   32
#define KVH_  8
#define HD_   128
#define KVDIM_ (KVH_ * HD_)          // 1024
#define NREP_ (NH_ / KVH_)           // 4
#define M_TOT (B_ * S_)              // 4096

typedef __attribute__((ext_vector_type(16))) _Float16     v16h;
typedef __attribute__((ext_vector_type(8)))  _Float16     v8h;
typedef __attribute__((ext_vector_type(8)))  float        v8f;
typedef __attribute__((ext_vector_type(4)))  unsigned int v4u;
typedef __attribute__((ext_vector_type(8)))  int          v8i;
typedef __attribute__((ext_vector_type(4)))  int          v4i;

__device__ __forceinline__ v8f wmma_f16(v16h a, v16h b, v8f c) {
    return __builtin_amdgcn_wmma_f32_16x16x32_f16(false, a, false, b,
                                                  (short)0, c, false, false);
}

// ---------------------------------------------------------------------------
// TDM: async DMA of a 2D f16 tile (tile_d0 contiguous elems x tile_d1 rows)
// from global [*, stride0] into LDS at byte offset lds_off. pad_ctrl packs
// {pad_enable, pad_interval, pad_amount} (D# group1 bits 20,24:22,31:25).
// D# layout per CDNA5 ISA. Completion: TENSORcnt.
// This toolchain's builtin is the 6-arg form:
//   (uint32x4 g0, int32x8 g1, int32x4 g2, int32x4 g3, int32x8 g4, i32 cpol)
// ---------------------------------------------------------------------------
__device__ __forceinline__ void tdm_load_2d(const void* gptr, unsigned lds_off,
                                            unsigned tensor_d0, unsigned tensor_d1,
                                            unsigned tile_d0, unsigned tile_d1,
                                            unsigned stride0, unsigned pad_ctrl) {
    unsigned long long ga = (unsigned long long)(uintptr_t)gptr;
    v4u g0;
    g0[0] = 1u;                                   // count=1, user descriptor
    g0[1] = lds_off;                              // lds_addr (bytes)
    g0[2] = (unsigned)ga;                         // global_addr[31:0]
    g0[3] = (unsigned)((ga >> 32) & 0x01ffffffu)  // global_addr[56:32]
          | (2u << 30);                           // type = 2 ("image")
    v8i g1;
    g1[0] = (int)((1u << 16) | pad_ctrl);                      // data_size=2B
    g1[1] = (int)(tensor_d0 << 16);                            // dim0[15:0]
    g1[2] = (int)((tensor_d0 >> 16) | (tensor_d1 << 16));      // dim0 hi|dim1 lo
    g1[3] = (int)((tensor_d1 >> 16) | (tile_d0 << 16));        // dim1 hi|tile0
    g1[4] = (int)(tile_d1 & 0xffffu);                          // tile1, tile2=0
    g1[5] = (int)stride0;                                      // dim0_stride lo32
    g1[6] = 0;
    g1[7] = 0;
    v4i z4 = {};
    v8i z8 = {};
    __builtin_amdgcn_tensor_load_to_lds(g0, g1, z4, z4, z8, 0);
}

// ---------------------------------------------------------------------------
// fp32 -> f16 conversion (n must be a multiple of 1024; all our sizes are)
// ---------------------------------------------------------------------------
__global__ __launch_bounds__(256) void cvt_f32_f16(const float* __restrict__ s,
                                                   _Float16* __restrict__ d,
                                                   int n) {
    int i = 4 * (blockIdx.x * 256 + threadIdx.x);
    if (i < n) {
        d[i + 0] = (_Float16)s[i + 0];
        d[i + 1] = (_Float16)s[i + 1];
        d[i + 2] = (_Float16)s[i + 2];
        d[i + 3] = (_Float16)s[i + 3];
    }
}

// ---------------------------------------------------------------------------
// In-place RoPE on f16 [M_TOT, 2^(ppr_shift+1)]; pairs interleaved (2i, 2i+1)
// within each 128-wide head; optional output scale (folds 1/sqrt(HD) into Q).
// ---------------------------------------------------------------------------
__global__ __launch_bounds__(256) void rope_scale(_Float16* __restrict__ x,
                                                  int ppr_shift,
                                                  float scale,
                                                  int total_pairs) {
    int idx = blockIdx.x * 256 + threadIdx.x;
    if (idx >= total_pairs) return;
    int row = idx >> ppr_shift;                 // b*S + s
    int p   = idx & ((1 << ppr_shift) - 1);     // pair index within row
    int i   = p & 63;                           // pair index within head
    int pos = row & (S_ - 1);                   // sequence position
    float ang = (float)pos * __expf(-(float)i * 0.14391156831212787f);
    float sn, cs;
    __sincosf(ang, &sn, &cs);
    _Float16* ptr = x + ((size_t)row << (ppr_shift + 1)) + 2 * p;
    float x0 = (float)ptr[0], x1 = (float)ptr[1];
    ptr[0] = (_Float16)((x0 * cs - x1 * sn) * scale);
    ptr[1] = (_Float16)((x0 * sn + x1 * cs) * scale);
}

// ---------------------------------------------------------------------------
// Tiled WMMA GEMM: C[M,N] = A[M,K] * B[K,N], A/B f16 row-major.
// 256 threads = 8 waves; tile 128x128x32; wave (wm,wn) owns 32x64.
// A tile DMA'd by the TDM (wave 0) with hardware row padding to LDA_=48 f16
// (96B rows => 32B-aligned v8h frag loads, bank-spread). B staged manually
// (transposed; TDM cannot transpose).
// ---------------------------------------------------------------------------
#define LDA_ 48
#define LDB_ 48
// pad_enable | pad_interval=3 (16 DWORDs = 64B) | pad_amount=7 (8 DWORDs = 32B)
#define PAD_A_ ((1u << 20) | (3u << 22) | (7u << 25))
template <typename OutT>
__global__ __launch_bounds__(256) void gemm_f16(const _Float16* __restrict__ A,
                                                const _Float16* __restrict__ B,
                                                OutT* __restrict__ C,
                                                int N, int K) {
    __shared__ _Float16 As[128 * LDA_];   // [row][k], TDM-padded rows
    __shared__ _Float16 Bs[128 * LDB_];   // transposed: [n][k]

    const int tid  = threadIdx.x;
    const int lane = tid & 31;
    const int wave = tid >> 5;
    const int wm   = wave >> 1;           // 0..3 (32 rows each)
    const int wn   = wave & 1;            // 0..1 (64 cols each)
    const int half = lane >> 4;
    const int l16  = lane & 15;
    const int m0   = blockIdx.y * 128;
    const int n0   = blockIdx.x * 128;

    const unsigned as_lds = (unsigned)(uintptr_t)(&As[0]);
    const _Float16* atile = A + (size_t)m0 * K;

    const int bkrow = tid >> 3, bseg = tid & 7;     // B staging assignment
    const _Float16* bgp = B + (size_t)bkrow * N + n0 + bseg * 16;

    v8f acc[2][4];
    #pragma unroll
    for (int i = 0; i < 2; ++i)
        #pragma unroll
        for (int j = 0; j < 4; ++j) acc[i][j] = (v8f)(0.0f);

    for (int k0 = 0; k0 < K; k0 += 32) {
        // Async DMA of the 128x32 A tile into padded LDS rows (one wave issues)
        if (wave == 0)
            tdm_load_2d(atile + k0, as_lds, (unsigned)K, 128u * 32u,
                        32u, 128u, (unsigned)K, PAD_A_);
        // Stage B tile transposed: coalesced row reads, b16 scatter into Bs[n][k]
        {
            v16h bv = *(const v16h*)(bgp + (size_t)k0 * N);
            #pragma unroll
            for (int e = 0; e < 16; ++e)
                Bs[(bseg * 16 + e) * LDB_ + bkrow] = bv[e];
        }
        if (k0 + 32 < K)
            __builtin_prefetch(bgp + (size_t)(k0 + 32) * N, 0, 1);
        if (wave == 0)
            __builtin_amdgcn_s_wait_tensorcnt(0);
        __syncthreads();

        // A fragments (16x32 f16): lane row = l16; K = half*16 + e (contiguous)
        v16h af[2];
        #pragma unroll
        for (int i = 0; i < 2; ++i) {
            int m = wm * 32 + i * 16 + l16;
            union { v16h v; v8h h2[2]; } u;
            u.h2[0] = *(const v8h*)(&As[m * LDA_ + half * 16]);
            u.h2[1] = *(const v8h*)(&As[m * LDA_ + half * 16 + 8]);
            af[i] = u.v;
        }
        // B fragments (32x16 f16): lane col = l16; K = half*16 + e (contiguous)
        v16h bf[4];
        #pragma unroll
        for (int j = 0; j < 4; ++j) {
            int n = wn * 64 + j * 16 + l16;
            bf[j] = *(const v16h*)(&Bs[n * LDB_ + half * 16]);
        }
        #pragma unroll
        for (int i = 0; i < 2; ++i)
            #pragma unroll
            for (int j = 0; j < 4; ++j)
                acc[i][j] = wmma_f16(af[i], bf[j], acc[i][j]);
        __syncthreads();
    }

    // Epilogue: C/D layout -> m = r + 8*half, n = l16
    #pragma unroll
    for (int i = 0; i < 2; ++i) {
        #pragma unroll
        for (int j = 0; j < 4; ++j) {
            OutT* cp = C + (size_t)(m0 + wm * 32 + i * 16 + half * 8) * N
                         + (n0 + wn * 64 + j * 16 + l16);
            #pragma unroll
            for (int r = 0; r < 8; ++r)
                cp[(size_t)r * N] = (OutT)acc[i][j][r];
        }
    }
}

// ---------------------------------------------------------------------------
// Flash attention, GQA causal. One wave per (b, head, 16-query tile).
// V tile DMA'd by the TDM at the top of each kv step; the DMA overlaps the
// score WMMAs and is only waited on right before the P.V fragment gathers.
// ---------------------------------------------------------------------------
#define LDP_ 40
__global__ __launch_bounds__(32) void attn(const _Float16* __restrict__ Q,
                                           const _Float16* __restrict__ Kb,
                                           const _Float16* __restrict__ Vb,
                                           _Float16* __restrict__ ctx) {
    __shared__ _Float16 P16[16 * LDP_];   // P in [m][k] for A-fragment reload
    __shared__ _Float16 Vs[32 * 128];     // V tile row-major [kv][hd]

    const int lane = threadIdx.x;
    const int half = lane >> 4;
    const int l16  = lane & 15;
    const int qt   = blockIdx.x;
    const int h    = blockIdx.y;
    const int b    = blockIdx.z;
    const int q0   = qt * 16;
    const int kvh  = h >> 2;              // NREP_ = 4

    const size_t qoff = (size_t)(b * S_) * DIM_   + (size_t)h   * HD_;
    const size_t koff = (size_t)(b * S_) * KVDIM_ + (size_t)kvh * HD_;
    const unsigned vs_lds = (unsigned)(uintptr_t)(&Vs[0]);

    // Q fragments: 4 chunks of 32 over HEAD_DIM
    v16h aq[4];
    #pragma unroll
    for (int c = 0; c < 4; ++c) {
        const _Float16* qr = Q + qoff + (size_t)(q0 + l16) * DIM_ + c * 32;
        union { v16h v; v8h h2[2]; } u;
        u.h2[0] = *(const v8h*)(qr + half * 16);
        u.h2[1] = *(const v8h*)(qr + half * 16 + 8);
        aq[c] = u.v;
    }

    float mrow[8], lrow[8];
    v8f acc[8];
    #pragma unroll
    for (int r = 0; r < 8; ++r) { mrow[r] = -1e30f; lrow[r] = 0.0f; }
    #pragma unroll
    for (int j = 0; j < 8; ++j) acc[j] = (v8f)(0.0f);

    const int kv_end = q0 + 16;
    for (int kv0 = 0; kv0 < kv_end; kv0 += 32) {
        // Kick off the async V-tile DMA (32 kv rows x 128 hd, row-major)
        tdm_load_2d(Vb + koff + (size_t)kv0 * KVDIM_, vs_lds,
                    (unsigned)KVDIM_, (unsigned)M_TOT,
                    (unsigned)HD_, 32u, (unsigned)KVDIM_, 0u);

        // Scores: hoist all 8 K fragments (contiguous 32B per lane), then WMMA
        v16h bk[8];
        #pragma unroll
        for (int c = 0; c < 4; ++c) {
            bk[c]     = *(const v16h*)(Kb + koff + (size_t)(kv0 + l16)      * KVDIM_ + c * 32 + half * 16);
            bk[4 + c] = *(const v16h*)(Kb + koff + (size_t)(kv0 + 16 + l16) * KVDIM_ + c * 32 + half * 16);
        }
        v8f c0 = (v8f)(0.0f), c1 = (v8f)(0.0f);
        #pragma unroll
        for (int c = 0; c < 4; ++c) {
            c0 = wmma_f16(aq[c], bk[c],     c0);
            c1 = wmma_f16(aq[c], bk[4 + c], c1);
        }

        // Online softmax per row (row = q0 + r + 8*half; col = kv0 + tile + l16)
        #pragma unroll
        for (int r = 0; r < 8; ++r) {
            int row = q0 + r + half * 8;
            float a0 = (kv0 + l16      <= row) ? c0[r] : -1e30f;
            float a1 = (kv0 + 16 + l16 <= row) ? c1[r] : -1e30f;
            float mx = fmaxf(a0, a1);
            mx = fmaxf(mx, __shfl_xor(mx, 1, 16));
            mx = fmaxf(mx, __shfl_xor(mx, 2, 16));
            mx = fmaxf(mx, __shfl_xor(mx, 4, 16));
            mx = fmaxf(mx, __shfl_xor(mx, 8, 16));
            float mnew  = fmaxf(mrow[r], mx);
            float alpha = __expf(mrow[r] - mnew);
            float p0 = __expf(a0 - mnew);
            float p1 = __expf(a1 - mnew);
            float rs = p0 + p1;
            rs += __shfl_xor(rs, 1, 16);
            rs += __shfl_xor(rs, 2, 16);
            rs += __shfl_xor(rs, 4, 16);
            rs += __shfl_xor(rs, 8, 16);
            lrow[r] = lrow[r] * alpha + rs;
            mrow[r] = mnew;
            #pragma unroll
            for (int j = 0; j < 8; ++j) acc[j][r] *= alpha;
            P16[(r + half * 8) * LDP_ + l16]      = (_Float16)p0;
            P16[(r + half * 8) * LDP_ + 16 + l16] = (_Float16)p1;
        }

        // Reload P as an A fragment (C-layout -> A-layout via LDS)
        union { v16h v; v8h h2[2]; } up;
        up.h2[0] = *(const v8h*)(&P16[l16 * LDP_ + half * 8]);
        up.h2[1] = *(const v8h*)(&P16[l16 * LDP_ + 16 + half * 8]);

        // V tile must be in LDS now
        __builtin_amdgcn_s_wait_tensorcnt(0);

        // O += P . V : B-frag gathered from LDS V tile
        #pragma unroll
        for (int j = 0; j < 8; ++j) {
            v16h bv;
            #pragma unroll
            for (int e = 0; e < 16; ++e)
                bv[e] = Vs[((e < 8 ? e : e + 8) + half * 8) * 128 + j * 16 + l16];
            acc[j] = wmma_f16(up.v, bv, acc[j]);
        }
    }

    // Normalize (one reciprocal per row) and store context (f16)
    float invl[8];
    #pragma unroll
    for (int r = 0; r < 8; ++r) invl[r] = 1.0f / lrow[r];

    _Float16* cp0 = ctx + (size_t)(b * S_ + q0 + half * 8) * DIM_ + h * HD_ + l16;
    #pragma unroll
    for (int j = 0; j < 8; ++j) {
        #pragma unroll
        for (int r = 0; r < 8; ++r)
            cp0[(size_t)r * DIM_ + j * 16] = (_Float16)(acc[j][r] * invl[r]);
    }
}

// ---------------------------------------------------------------------------
// Launch
// ---------------------------------------------------------------------------
extern "C" void kernel_launch(void* const* d_in, const int* in_sizes, int n_in,
                              void* d_out, int out_size, void* d_ws, size_t ws_size,
                              hipStream_t stream) {
    const float* x  = (const float*)d_in[0];
    const float* Wq = (const float*)d_in[1];
    const float* Wk = (const float*)d_in[2];
    const float* Wv = (const float*)d_in[3];
    const float* Wo = (const float*)d_in[4];
    float* out = (float*)d_out;

    char*  ws  = (char*)d_ws;
    size_t off = 0;
    auto carve = [&](size_t n_elem) -> _Float16* {
        _Float16* p = (_Float16*)(ws + off);
        off += ((n_elem * sizeof(_Float16)) + 255) & ~(size_t)255;
        return p;
    };
    _Float16* xh  = carve((size_t)M_TOT * DIM_);
    _Float16* Wqh = carve((size_t)DIM_ * DIM_);
    _Float16* Wkh = carve((size_t)DIM_ * KVDIM_);
    _Float16* Wvh = carve((size_t)DIM_ * KVDIM_);
    _Float16* Woh = carve((size_t)DIM_ * DIM_);
    _Float16* Qb  = carve((size_t)M_TOT * DIM_);
    _Float16* Kb  = carve((size_t)M_TOT * KVDIM_);
    _Float16* Vb  = carve((size_t)M_TOT * KVDIM_);
    _Float16* ctx = carve((size_t)M_TOT * DIM_);

    auto cvt = [&](const float* s, _Float16* d, size_t n) {
        cvt_f32_f16<<<dim3((unsigned)(n / 1024)), 256, 0, stream>>>(s, d, (int)n);
    };
    cvt(x,  xh,  (size_t)M_TOT * DIM_);
    cvt(Wq, Wqh, (size_t)DIM_ * DIM_);
    cvt(Wk, Wkh, (size_t)DIM_ * KVDIM_);
    cvt(Wv, Wvh, (size_t)DIM_ * KVDIM_);
    cvt(Wo, Woh, (size_t)DIM_ * DIM_);

    // Projections (f16 out)
    gemm_f16<_Float16><<<dim3(DIM_ / 128,   M_TOT / 128), 256, 0, stream>>>(xh, Wqh, Qb, DIM_,   DIM_);
    gemm_f16<_Float16><<<dim3(KVDIM_ / 128, M_TOT / 128), 256, 0, stream>>>(xh, Wkh, Kb, KVDIM_, DIM_);
    gemm_f16<_Float16><<<dim3(KVDIM_ / 128, M_TOT / 128), 256, 0, stream>>>(xh, Wvh, Vb, KVDIM_, DIM_);

    // RoPE (Q also picks up the 1/sqrt(HEAD_DIM) score scale)
    {
        int tpq = M_TOT * (DIM_ / 2);      // pairs/row = 2048 = 2^11
        rope_scale<<<dim3((tpq + 255) / 256), 256, 0, stream>>>(Qb, 11, 0.08838834764831845f, tpq);
        int tpk = M_TOT * (KVDIM_ / 2);    // pairs/row = 512 = 2^9
        rope_scale<<<dim3((tpk + 255) / 256), 256, 0, stream>>>(Kb, 9, 1.0f, tpk);
    }

    // Flash attention: one wave per (b, head, 16-query tile)
    attn<<<dim3(S_ / 16, NH_, B_), 32, 0, stream>>>(Qb, Kb, Vb, ctx);

    // Output projection -> fp32 d_out
    gemm_f16<float><<<dim3(DIM_ / 128, M_TOT / 128), 256, 0, stream>>>(ctx, Woh, out, DIM_, DIM_);
}
